// AttentionRNN_4922032521714
// MI455X (gfx1250) — matrix operand order repaired and checked by reading, not compiled, not run on hardware
//
#include <hip/hip_runtime.h>
#include <math.h>

// ---------------------------------------------------------------------------
// AttentionRNN for MI455X (gfx1250, wave32, WMMA bf16, async-to-LDS staging)
//   VOCAB=32000, H=512, B=16, S=512
// Pipeline (all on `stream`):
//   0) convert Wxh/Whh/fc_w fp32 -> bf16 (workspace)
//   1) embedding gather -> xe bf16 [8192,512]
//   2) xproj = xe @ Wxh^T + b      (WMMA GEMM, async LDS double-buffer)
//   3) RNN scan (1 WG, h state in LDS, WMMA per step) -> combined[:, :512] bf16
//   4) causal attention per (b, 16-row tile)          -> combined[:, 512:] bf16
//   5) out = combined @ fc_w^T + fc_b (WMMA GEMM, fp32 out [8192,32000])
// ---------------------------------------------------------------------------

typedef __attribute__((ext_vector_type(16))) __bf16 v16bf;
typedef __attribute__((ext_vector_type(8)))  float  v8f;
typedef __attribute__((ext_vector_type(4)))  int    v4i;

#define HDIM   512
#define BATCH  16
#define SEQ    512
#define VOCABN 32000
#define ROWS   (BATCH * SEQ)   // 8192
#define KFC    (2 * HDIM)      // 1024

// ---- CDNA5 async global->LDS path (guarded; falls back to direct copies) ---
#if defined(__has_builtin)
#if __has_builtin(__builtin_amdgcn_global_load_async_to_lds_b128) && \
    __has_builtin(__builtin_amdgcn_s_wait_asynccnt)
#define USE_ASYNC_LDS 1
#endif
#endif
#ifndef USE_ASYNC_LDS
#define USE_ASYNC_LDS 0
#endif

#if USE_ASYNC_LDS
#warning "CDNA5: global_load_async_to_lds builtins AVAILABLE - async staging enabled"
// addrspace-qualified pointer types (attribute binds reliably via typedef)
typedef __attribute__((address_space(1))) v4i* gv4i_ptr;  // global
typedef __attribute__((address_space(3))) v4i* lv4i_ptr;  // LDS (32-bit)
#else
#warning "CDNA5: async-to-LDS builtins NOT available - using direct LDS copies"
#endif

// copy 16 bytes global -> LDS (8 bf16), async when available
__device__ __forceinline__ void cp16_g2l(const __bf16* g, __bf16* l) {
#if USE_ASYNC_LDS
  // generic global addr == AS1 addr numerically; generic LDS addr low 32 bits
  // == AS3 offset
  gv4i_ptr gp = (gv4i_ptr)(unsigned long long)g;
  lv4i_ptr lp = (lv4i_ptr)(unsigned int)(unsigned long long)l;
  __builtin_amdgcn_global_load_async_to_lds_b128(gp, lp, 0, 0);
#else
  *(uint4*)l = *(const uint4*)g;
#endif
}
__device__ __forceinline__ void wait_async_2() {
#if USE_ASYNC_LDS
  __builtin_amdgcn_s_wait_asynccnt(2);
#endif
}
__device__ __forceinline__ void wait_async_0() {
#if USE_ASYNC_LDS
  __builtin_amdgcn_s_wait_asynccnt(0);
#endif
}

__device__ __forceinline__ v8f wmma_bf16(v16bf a, v16bf b, v8f c) {
  return __builtin_amdgcn_wmma_f32_16x16x32_bf16(
      false, a, false, b, (short)0, c, false, false);
}

// A fragment: 16x32 bf16 tile, row-major source with leading dim ld.
// Contiguous k order used here, not the ISA table order (16-bit A 16x32): lane = {m = L&15, g = L>>4};
//   VGPR0..3: K = g*16 + {0..7};  VGPR4..7: K = g*16 + 8 + {0..7}, K-pairs packed.
__device__ __forceinline__ v16bf load_a16x32(const __bf16* base, int ld) {
  const int lane = threadIdx.x & 31;
  const int m = lane & 15, g = lane >> 4;
  const __bf16* row = base + (size_t)m * ld;
  v16bf f;
#pragma unroll
  for (int j = 0; j < 4; ++j) {
    int k = g * 16 + 2 * j;
    f[2 * j]     = row[k];
    f[2 * j + 1] = row[k + 1];
  }
#pragma unroll
  for (int j = 0; j < 4; ++j) {
    int k = g * 16 + 8 + 2 * j;
    f[8 + 2 * j]     = row[k];
    f[8 + 2 * j + 1] = row[k + 1];
  }
  return f;
}

// B fragment (32x16, B = W^T) from row-major W[N,K] with leading dim ld.
// lane = {n = L&15, g = L>>4}; K = g*16 + {0..15} packed sequentially.
__device__ __forceinline__ v16bf load_b_nk(const __bf16* base, int ld) {
  const int lane = threadIdx.x & 31;
  const int n = lane & 15, g = lane >> 4;
  const __bf16* row = base + (size_t)n * ld + g * 16;
  v16bf f;
#pragma unroll
  for (int j = 0; j < 16; ++j) f[j] = row[j];
  return f;
}

// B fragment (32x16) from row-major B[K,N] with leading dim ld (strided).
__device__ __forceinline__ v16bf load_b_kn(const __bf16* base, int ld) {
  const int lane = threadIdx.x & 31;
  const int n = lane & 15, g = lane >> 4;
  const __bf16* col = base + n + (size_t)(g * 16) * ld;
  v16bf f;
#pragma unroll
  for (int j = 0; j < 16; ++j) f[j] = col[(size_t)j * ld];
  return f;
}

// ---------------------------------------------------------------------------
// 0) fp32 -> bf16 conversion (grid-stride)
// ---------------------------------------------------------------------------
__global__ void k_f32_to_bf16(const float* __restrict__ src,
                              __bf16* __restrict__ dst, int n) {
  int i = blockIdx.x * blockDim.x + threadIdx.x;
  int stride = gridDim.x * blockDim.x;
  for (; i < n; i += stride) dst[i] = (__bf16)src[i];
}

// ---------------------------------------------------------------------------
// 1) embedding gather: xe[row][:] = bf16(emb[x[row]][:])
// ---------------------------------------------------------------------------
__global__ void k_embed(const int* __restrict__ x,
                        const float* __restrict__ emb,
                        __bf16* __restrict__ xe) {
  const int row = blockIdx.x;                 // b*S + s
  const int idx = x[row];
  const float* e = emb + (size_t)idx * HDIM;
  __bf16* o = xe + (size_t)row * HDIM;
  for (int i = threadIdx.x; i < HDIM; i += blockDim.x) o[i] = (__bf16)e[i];
}

// ---------------------------------------------------------------------------
// Generic WMMA GEMM: C[M,N] = A_bf16[M,K] @ W_bf16[N,K]^T + bias[N]
// block = 256 threads (8 waves) -> 64x64 tile; wave = 16x32 (two 16x16 accs).
// A/W K-panels staged through LDS, double-buffered with async-to-LDS loads.
// M % 64 == 0, N % 64 == 0, K % 32 == 0 (true for all uses here).
// ---------------------------------------------------------------------------
__global__ __launch_bounds__(256) void k_gemm_wmma(
    const __bf16* __restrict__ A, int lda,
    const __bf16* __restrict__ W, int ldw,
    const float* __restrict__ bias,
    float* __restrict__ C, int ldc, int K) {
  __shared__ __bf16 Abuf[2][64][32];   // 2 x 4 KB
  __shared__ __bf16 Wbuf[2][64][32];   // 2 x 4 KB

  const int tid = threadIdx.x;
  const int w = tid >> 5;
  const int m_off = blockIdx.y * 64 + (w & 3) * 16;
  const int n_off = blockIdx.x * 64 + (w >> 2) * 32;

  // per-thread staging assignment: one 16B chunk of A and W per K-step
  const int srow = tid >> 2;           // 0..63
  const int sch  = (tid & 3) * 8;      // 0,8,16,24 (bf16 elems)
  const __bf16* Ag = A + (size_t)(blockIdx.y * 64 + srow) * lda + sch;
  const __bf16* Wg = W + (size_t)(blockIdx.x * 64 + srow) * ldw + sch;

  v8f acc0 = {};
  v8f acc1 = {};
  const int nk = K >> 5;

  // prologue: stage K-step 0 into buffer 0
  cp16_g2l(Ag, &Abuf[0][srow][sch]);
  cp16_g2l(Wg, &Wbuf[0][srow][sch]);

  for (int k = 0; k < nk; ++k) {
    const int cur = k & 1;
    if (k + 1 < nk) {
      const int kk1 = (k + 1) << 5;
      if (k + 2 < nk) {  // keep next+1 panel warm (global_prefetch_b8, near)
        __builtin_prefetch(Ag + kk1 + 32, 0, 3);
        __builtin_prefetch(Wg + kk1 + 32, 0, 3);
      }
      cp16_g2l(Ag + kk1, &Abuf[cur ^ 1][srow][sch]);
      cp16_g2l(Wg + kk1, &Wbuf[cur ^ 1][srow][sch]);
      wait_async_2();                  // own stage-k loads complete
    } else {
      wait_async_0();
    }
    __syncthreads();                   // everyone's stage-k tile in LDS

    v16bf a  = load_a16x32(&Abuf[cur][(w & 3) * 16][0], 32);
    v16bf b0 = load_b_nk(&Wbuf[cur][(w >> 2) * 32][0], 32);
    v16bf b1 = load_b_nk(&Wbuf[cur][(w >> 2) * 32 + 16][0], 32);
    acc0 = wmma_bf16(a, b0, acc0);
    acc1 = wmma_bf16(a, b1, acc1);

    __syncthreads();                   // reads done before this buf is re-staged
  }

  const int lane = tid & 31;
  const int n = lane & 15, mb = (lane >> 4) * 8;
#pragma unroll
  for (int j = 0; j < 8; ++j) {
    const int r = m_off + mb + j;
    const int c0 = n_off + n, c1 = n_off + 16 + n;
    C[(size_t)r * ldc + c0] = acc0[j] + bias[c0];
    C[(size_t)r * ldc + c1] = acc1[j] + bias[c1];
  }
}

// ---------------------------------------------------------------------------
// 3) RNN scan: h_{t} = tanh(xproj_t + h_{t-1} @ Whh^T + Whh_b)
// One workgroup, 512 threads = 16 waves; wave w owns output cols [32w, 32w+32).
// h state double-buffered in LDS as bf16 (A operand). hs -> combined[:, :512].
// ---------------------------------------------------------------------------
__global__ __launch_bounds__(512) void k_rnn_scan(
    const float* __restrict__ xproj,      // [8192,512] row = b*S + t
    const __bf16* __restrict__ Whh,       // [512,512] bf16
    const float* __restrict__ Whh_b,      // [512]
    __bf16* __restrict__ combined) {      // [8192,1024], cols 0..511
  __shared__ __bf16 hbuf[2][BATCH][HDIM]; // 2 * 16KB

  const int tid = threadIdx.x;
  for (int i = tid; i < BATCH * HDIM; i += blockDim.x)
    (&hbuf[0][0][0])[i] = (__bf16)0.0f;
  __syncthreads();

  const int w = tid >> 5, lane = tid & 31;
  const int n_off = w * 32;
  const int n = lane & 15, mb = (lane >> 4) * 8;
  int cur = 0;

  for (int t = 0; t < SEQ; ++t) {
    v8f acc0 = {};
    v8f acc1 = {};
    const __bf16* hb = &hbuf[cur][0][0];
#pragma unroll 4
    for (int kk = 0; kk < HDIM; kk += 32) {
      v16bf a  = load_a16x32(hb + kk, HDIM);
      v16bf b0 = load_b_nk(Whh + (size_t)n_off * HDIM + kk, HDIM);
      v16bf b1 = load_b_nk(Whh + (size_t)(n_off + 16) * HDIM + kk, HDIM);
      acc0 = wmma_bf16(a, b0, acc0);
      acc1 = wmma_bf16(a, b1, acc1);
    }
#pragma unroll
    for (int j = 0; j < 8; ++j) {
      const int b = mb + j;                      // batch index (M=16 exactly)
      const int c0 = n_off + n, c1 = n_off + 16 + n;
      const size_t row = (size_t)b * SEQ + t;
      float v0 = tanhf(acc0[j] + xproj[row * HDIM + c0] + Whh_b[c0]);
      float v1 = tanhf(acc1[j] + xproj[row * HDIM + c1] + Whh_b[c1]);
      hbuf[cur ^ 1][b][c0] = (__bf16)v0;
      hbuf[cur ^ 1][b][c1] = (__bf16)v1;
      combined[row * KFC + c0] = (__bf16)v0;
      combined[row * KFC + c1] = (__bf16)v1;
    }
    __syncthreads();
    cur ^= 1;
  }
}

// ---------------------------------------------------------------------------
// 4) causal attention, one block per (batch, 16-query tile): 512 blocks.
//   scores = q @ hs^T (WMMA) -> mask -> softmax (shfl within 16-lane groups)
//   context = softmax @ hs (WMMA) -> combined[:, 512:1024]
// ---------------------------------------------------------------------------
__global__ __launch_bounds__(256) void k_attn(__bf16* __restrict__ combined) {
  __shared__ float  sc[16][SEQ];   // 32 KB
  __shared__ __bf16 wt[16][SEQ];   // 16 KB

  const int blk = blockIdx.x;         // 0..511
  const int b  = blk >> 5;            // batch
  const int t0 = (blk & 31) * 16;     // query-row tile base
  const __bf16* base = combined + (size_t)b * SEQ * KFC; // hs at col 0, ld=KFC

  const int tid = threadIdx.x, w = tid >> 5, lane = tid & 31;
  const int n = lane & 15, mb = (lane >> 4) * 8;

  // ---- phase 1: scores[16][512] ----
  for (int sblk = w * 64; sblk < w * 64 + 64; sblk += 16) {
    v8f acc = {};
    for (int kk = 0; kk < HDIM; kk += 32) {
      v16bf a  = load_a16x32(base + (size_t)t0 * KFC + kk, KFC);
      v16bf bb = load_b_nk(base + (size_t)sblk * KFC + kk, KFC);
      acc = wmma_bf16(a, bb, acc);
    }
#pragma unroll
    for (int j = 0; j < 8; ++j) sc[mb + j][sblk + n] = acc[j];
  }
  __syncthreads();

  // ---- phase 2: masked softmax; row = tid>>4, 16 lanes per row ----
  {
    const int row = tid >> 4;        // 0..15 (two rows per wave: lanes 0-15/16-31)
    const int c = tid & 15;
    const int tglob = t0 + row;
    float mx = -3.0e38f;
    for (int s = c; s < SEQ; s += 16) {
      float v = (s <= tglob) ? sc[row][s] : -1.0e30f;
      sc[row][s] = v;
      mx = fmaxf(mx, v);
    }
#pragma unroll
    for (int off = 8; off; off >>= 1) mx = fmaxf(mx, __shfl_xor(mx, off, 32));
    float sum = 0.0f;
    for (int s = c; s < SEQ; s += 16) {
      float e = __expf(sc[row][s] - mx);
      sc[row][s] = e;
      sum += e;
    }
#pragma unroll
    for (int off = 8; off; off >>= 1) sum += __shfl_xor(sum, off, 32);
    const float inv = 1.0f / sum;
    for (int s = c; s < SEQ; s += 16) wt[row][s] = (__bf16)(sc[row][s] * inv);
  }
  __syncthreads();

  // ---- phase 3: context = wt[16,512] @ hs[512,512] ----
  for (int hblk = w * 64; hblk < w * 64 + 64; hblk += 16) {
    v8f acc = {};
    for (int kk = 0; kk < SEQ; kk += 32) {
      v16bf a  = load_a16x32(&wt[0][0] + kk, SEQ);
      v16bf bb = load_b_kn(base + (size_t)kk * KFC + hblk, KFC); // hs is [K,N]
      acc = wmma_bf16(a, bb, acc);
    }
#pragma unroll
    for (int j = 0; j < 8; ++j)
      combined[((size_t)b * SEQ + t0 + mb + j) * KFC + HDIM + hblk + n] =
          (__bf16)acc[j];
  }
}

// ---------------------------------------------------------------------------
// launch
// ---------------------------------------------------------------------------
extern "C" void kernel_launch(void* const* d_in, const int* in_sizes, int n_in,
                              void* d_out, int out_size, void* d_ws, size_t ws_size,
                              hipStream_t stream) {
  const int*   x     = (const int*)  d_in[0];   // [B,S]
  const float* emb   = (const float*)d_in[1];   // [V,H]
  const float* Wxh_w = (const float*)d_in[2];   // [H,H]
  const float* Wxh_b = (const float*)d_in[3];   // [H]
  const float* Whh_w = (const float*)d_in[4];   // [H,H]
  const float* Whh_b = (const float*)d_in[5];   // [H]
  const float* fc_w  = (const float*)d_in[6];   // [V,2H]
  const float* fc_b  = (const float*)d_in[7];   // [V]
  float* out = (float*)d_out;                   // [B,S,V]

  // workspace layout (bytes), 256B aligned
  char* ws = (char*)d_ws;
  size_t off = 0;
  auto alloc = [&](size_t bytes) {
    char* p = ws + off;
    off = (off + bytes + 255) & ~(size_t)255;
    return p;
  };
  __bf16* Wxh_bf = (__bf16*)alloc((size_t)HDIM * HDIM * 2);        // 512 KB
  __bf16* Whh_bf = (__bf16*)alloc((size_t)HDIM * HDIM * 2);        // 512 KB
  __bf16* fcw_bf = (__bf16*)alloc((size_t)VOCABN * KFC * 2);       // 64 MB
  __bf16* xe_bf  = (__bf16*)alloc((size_t)ROWS * HDIM * 2);        // 8 MB
  float*  xproj  = (float*) alloc((size_t)ROWS * HDIM * 4);        // 16 MB
  __bf16* comb   = (__bf16*)alloc((size_t)ROWS * KFC * 2);         // 16 MB
  (void)ws_size; (void)n_in; (void)in_sizes; (void)out_size;

  // 0) weight conversions
  k_f32_to_bf16<<<512,  256, 0, stream>>>(Wxh_w, Wxh_bf, HDIM * HDIM);
  k_f32_to_bf16<<<512,  256, 0, stream>>>(Whh_w, Whh_bf, HDIM * HDIM);
  k_f32_to_bf16<<<4096, 256, 0, stream>>>(fc_w,  fcw_bf, VOCABN * KFC);

  // 1) embedding gather
  k_embed<<<ROWS, 256, 0, stream>>>(x, emb, xe_bf);

  // 2) xproj = xe @ Wxh^T + Wxh_b   (M=8192, N=512, K=512)
  {
    dim3 grid(HDIM / 64, ROWS / 64);
    k_gemm_wmma<<<grid, 256, 0, stream>>>(xe_bf, HDIM, Wxh_bf, HDIM,
                                          Wxh_b, xproj, HDIM, HDIM);
  }

  // 3) RNN scan -> combined[:, :512]
  k_rnn_scan<<<1, 512, 0, stream>>>(xproj, Whh_bf, Whh_b, comb);

  // 4) attention -> combined[:, 512:]
  k_attn<<<BATCH * (SEQ / 16), 256, 0, stream>>>(comb);

  // 5) out = combined @ fc_w^T + fc_b   (M=8192, N=32000, K=1024)
  {
    dim3 grid(VOCABN / 64, ROWS / 64);
    k_gemm_wmma<<<grid, 256, 0, stream>>>(comb, KFC, fcw_bf, KFC,
                                          fc_b, out, VOCABN, KFC);
  }
}
